// AttentionLayer_45372034515380
// MI455X (gfx1250) — matrix operand order repaired and checked by reading, not compiled, not run on hardware
//
#include <hip/hip_runtime.h>
#include <hip/hip_bf16.h>

typedef __attribute__((ext_vector_type(16))) __bf16 v16bf;
typedef __attribute__((ext_vector_type(8)))  __bf16 v8bf;
typedef __attribute__((ext_vector_type(2)))  __bf16 v2bf;
typedef __attribute__((ext_vector_type(8)))  float  v8f;
typedef __attribute__((ext_vector_type(4)))  float  v4f;

union BF16x16 { v16bf v; v8bf h[2]; __bf16 e[16]; unsigned u[8]; };
union BF16x8  { v8bf v; __bf16 e[8]; unsigned u[4]; };

__device__ __forceinline__ __bf16 f2bf(float f) { return (__bf16)f; }

__device__ __forceinline__ unsigned pk_bf16(float x, float y) {
  v2bf r; r[0] = (__bf16)x; r[1] = (__bf16)y;
  unsigned u; __builtin_memcpy(&u, &r, 4); return u;
}

// Wave-local LDS fence: one wave per workgroup, lanes are lockstep, DS ops are
// in-order per wave -> cross-lane LDS visibility needs only s_wait_dscnt 0.
// Unlike __syncthreads() this does NOT drain LOADcnt, so pipelined global
// loads stay in flight across the softmax.
__device__ __forceinline__ void wave_lds_fence() {
  asm volatile("s_wait_dscnt 0x0" ::: "memory");
}

#define WMMA_BF16(A, Bm, C) \
  __builtin_amdgcn_wmma_f32_16x16x32_bf16(false, (A), false, (Bm), (short)0, (C), false, false)

// ---------------------------------------------------------------------------
// Weight transpose + bf16 convert: dst[batch][col][row] = bf16(src[batch][row][col])
// ---------------------------------------------------------------------------
__global__ void transpose_w_kernel(const float* __restrict__ src,
                                   __bf16* __restrict__ dst,
                                   int nb, int nr, int nc) {
  size_t i = (size_t)blockIdx.x * blockDim.x + threadIdx.x;
  size_t total = (size_t)nb * nr * nc;
  if (i >= total) return;
  int c = (int)(i % nc);
  size_t t = i / nc;
  int r = (int)(t % nr);
  int bb = (int)(t / nr);
  dst[((size_t)bb * nc + c) * nr + r] = f2bf(src[i]);
}

// ---------------------------------------------------------------------------
// Per-head projection: out = X(BT,D) * W_h + bias. Wave = 16x64 (t x p) strip.
// Double-buffered K-loop (ping-pong, statically indexed). TRANS=1 stores (b,h,p,t).
// ---------------------------------------------------------------------------
template <int TRANS>
__global__ __launch_bounds__(256) void proj_kernel(
    const float* __restrict__ X,     // (B*T, D) fp32
    const __bf16* __restrict__ WT,   // (H, P, D) bf16 pre-transposed
    const float* __restrict__ bias,  // (H*P)
    __bf16* __restrict__ out) {
  const int T = 2048, D = 1024, H = 16, P = 64;
  const int lane = threadIdx.x & 31;
  const int wid  = blockIdx.x * (blockDim.x >> 5) + (threadIdx.x >> 5);
  const int tTiles = T / 16;
  const int qt = wid % tTiles;
  const int h  = (wid / tTiles) % H;
  const int b  = wid / (tTiles * H);

  const int m  = lane & 15;
  const int hl = lane >> 4;
  const int n  = lane & 15;

  const float*  xrow = X  + (size_t)(b * T + qt * 16 + m) * D;
  const __bf16* wn   = WT + ((size_t)h * P + n) * D;

  auto loadA = [&](v4f* f, int kk) {
    const float* p0 = xrow + kk + hl * 8;
    const float* p1 = xrow + kk + 16 + hl * 8;
    f[0] = *(const v4f*)(p0);
    f[1] = *(const v4f*)(p0 + 4);
    f[2] = *(const v4f*)(p1);
    f[3] = *(const v4f*)(p1 + 4);
  };
  auto loadB = [&](BF16x16* dst, int kk) {
#pragma unroll
    for (int j = 0; j < 4; ++j) {
      const __bf16* wp = wn + (size_t)j * 16 * D + kk + hl * 8;
      dst[j].h[0] = *(const v8bf*)(wp);
      dst[j].h[1] = *(const v8bf*)(wp + 16);
    }
  };
  auto cvtA = [&](const v4f* f) {
    BF16x16 a;
#pragma unroll
    for (int q = 0; q < 4; ++q) {
      a.u[2 * q]     = pk_bf16(f[q][0], f[q][1]);
      a.u[2 * q + 1] = pk_bf16(f[q][2], f[q][3]);
    }
    return a;
  };

  v8f acc[4] = {};
  v4f fa[4];
  BF16x16 b0[4], b1[4];
  loadA(fa, 0);
  loadB(b0, 0);
  for (int kk = 0; kk < D; kk += 64) {
    // phase 0: compute kk with b0; fetch kk+32 into b1
    BF16x16 a0 = cvtA(fa);
    loadA(fa, kk + 32);
    loadB(b1, kk + 32);
#pragma unroll
    for (int j = 0; j < 4; ++j) acc[j] = WMMA_BF16(a0.v, b0[j].v, acc[j]);
    // phase 1: compute kk+32 with b1; fetch kk+64 into b0
    BF16x16 a1 = cvtA(fa);
    if (kk + 64 < D) {
      loadA(fa, kk + 64);
      loadB(b0, kk + 64);
    }
#pragma unroll
    for (int j = 0; j < 4; ++j) acc[j] = WMMA_BF16(a1.v, b1[j].v, acc[j]);
  }

  // bias + store
#pragma unroll
  for (int j = 0; j < 4; ++j) {
    const int p = j * 16 + n;
    const float bv = bias[h * 64 + p];
    if (TRANS) {
      // rows are contiguous in memory -> one b128 store of 8 bf16
      BF16x8 o;
#pragma unroll
      for (int r2 = 0; r2 < 4; ++r2)
        o.u[r2] = pk_bf16(acc[j][2 * r2] + bv, acc[j][2 * r2 + 1] + bv);
      *(v8bf*)(out + (((size_t)(b * H + h)) * P + p) * T + qt * 16 + 8 * hl) = o.v;
    } else {
#pragma unroll
      for (int r = 0; r < 8; ++r) {
        const int row = qt * 16 + r + 8 * hl;
        out[(((size_t)(b * H + h)) * T + row) * P + p] = f2bf(acc[j][r] + bv);
      }
    }
  }
}

// ---------------------------------------------------------------------------
// Flash attention. ONE WAVE PER WORKGROUP. Cross-lane LDS ordering uses
// s_wait_dscnt-only fences (split CDNA5 counters) so the software-pipelined
// K/V global loads stay in flight across the entire softmax.
// ---------------------------------------------------------------------------
__global__ __launch_bounds__(32) void attn_kernel(
    const __bf16* __restrict__ qh,   // (B,H,T,P)
    const __bf16* __restrict__ kh,   // (B,H,T,P)
    const __bf16* __restrict__ vhT,  // (B,H,P,T)
    const float* __restrict__ mask,  // (T,T)
    __bf16* __restrict__ ctx) {      // (B,T,H*P)
  const int T = 2048, P = 64, H = 16;
  __shared__ float  Ssl[16 * 32];
  __shared__ __bf16 Psl[16 * 32];
  __shared__ float  Csl[16];

  const int lane = threadIdx.x & 31;
  const int wid  = blockIdx.x;
  const int qTiles = T / 16;
  const int qt = wid % qTiles;
  const int h  = (wid / qTiles) % H;
  const int b  = wid / (qTiles * H);

  const int m  = lane & 15;
  const int hl = lane >> 4;
  const int n  = lane & 15;
  const size_t bh = (size_t)(b * H + h);

  // Q A-fragments live in registers for the whole pass
  const __bf16* qbase = qh + (bh * T + qt * 16 + m) * P;
  BF16x16 aq0, aq1;
  aq0.h[0] = *(const v8bf*)(qbase + hl * 8);
  aq0.h[1] = *(const v8bf*)(qbase + 16 + hl * 8);
  aq1.h[0] = *(const v8bf*)(qbase + 32 + hl * 8);
  aq1.h[1] = *(const v8bf*)(qbase + 48 + hl * 8);

  v8f acc[4] = {};
  float M = -1e30f, l = 0.0f;      // lanes 0..15 own row = lane
  const float scale = 0.125f;      // 1/sqrt(64)

  const __bf16* krow = kh + (bh * T + n) * P;        // + (kt + sj*16)*P
  const __bf16* vb   = vhT + (bh * P + n) * T;       // + j*16*T + kt
  const float*  mrow0 = mask + (size_t)(qt * 16 + lane) * T;

  auto loadK = [&](BF16x16 (&k0)[2], BF16x16 (&k1)[2], int kt) {
#pragma unroll
    for (int sj = 0; sj < 2; ++sj) {
      const __bf16* kb = krow + (size_t)(kt + sj * 16) * P;
      k0[sj].h[0] = *(const v8bf*)(kb + hl * 8);
      k0[sj].h[1] = *(const v8bf*)(kb + 16 + hl * 8);
      k1[sj].h[0] = *(const v8bf*)(kb + 32 + hl * 8);
      k1[sj].h[1] = *(const v8bf*)(kb + 48 + hl * 8);
    }
  };
  auto loadV = [&](BF16x16 (&bvf)[4], int kt) {
#pragma unroll
    for (int j = 0; j < 4; ++j) {
      const __bf16* vp = vb + (size_t)j * 16 * T + kt + hl * 8;
      bvf[j].h[0] = *(const v8bf*)(vp);
      bvf[j].h[1] = *(const v8bf*)(vp + 16);
    }
  };

  auto step = [&](int kt, BF16x16 (&kc0)[2], BF16x16 (&kc1)[2],
                  BF16x16 (&kn0)[2], BF16x16 (&kn1)[2], bool loadNext) {
    // speculative prefetch two tiles ahead (next tile is a real load already)
    __builtin_prefetch(krow + (size_t)(kt + 64) * P, 0, 1);
    __builtin_prefetch(vb + kt + 64, 0, 1);

    // ---- scores with the pre-loaded K fragments ----
#pragma unroll
    for (int sj = 0; sj < 2; ++sj) {
      v8f s = {};
      s = WMMA_BF16(aq0.v, kc0[sj].v, s);
      s = WMMA_BF16(aq1.v, kc1[sj].v, s);
#pragma unroll
      for (int r = 0; r < 8; ++r)
        Ssl[(r + 8 * hl) * 32 + sj * 16 + n] = s[r];
    }

    // issue next-tile K loads and this-tile V loads; they stay in flight
    // through the softmax (fence below drains DScnt only, not LOADcnt)
    if (loadNext) loadK(kn0, kn1, kt + 32);
    BF16x16 bvf[4];
    loadV(bvf, kt);

    wave_lds_fence();

    // ---- streaming softmax: lane < 16 owns one row ----
    if (lane < 16) {
      const float* mrow = mrow0 + kt;
      float newmax = M;
#pragma unroll
      for (int j = 0; j < 32; ++j) {
        float v = Ssl[lane * 32 + j] * scale + (1.0f - mrow[j]) * (-1e9f);
        Ssl[lane * 32 + j] = v;          // write masked score back (saves 32 VGPRs)
        newmax = fmaxf(newmax, v);
      }
      float c = __expf(M - newmax);
      float rs = 0.0f;
#pragma unroll
      for (int j = 0; j < 16; ++j) {
        float e0 = __expf(Ssl[lane * 32 + 2 * j] - newmax);
        float e1 = __expf(Ssl[lane * 32 + 2 * j + 1] - newmax);
        rs += e0 + e1;
        *(unsigned*)&Psl[lane * 32 + 2 * j] = pk_bf16(e0, e1);
      }
      l = l * c + rs;
      M = newmax;
      Csl[lane] = c;
    }
    wave_lds_fence();

    // ---- rescale + PV (V fragments already resident) ----
    float cf[8];
#pragma unroll
    for (int r = 0; r < 8; ++r) cf[r] = Csl[r + 8 * hl];
#pragma unroll
    for (int j = 0; j < 4; ++j)
#pragma unroll
      for (int r = 0; r < 8; ++r) acc[j][r] *= cf[r];
    BF16x16 ap;
    ap.h[0] = *(const v8bf*)(Psl + m * 32 + hl * 8);
    ap.h[1] = *(const v8bf*)(Psl + m * 32 + 16 + hl * 8);
#pragma unroll
    for (int j = 0; j < 4; ++j) acc[j] = WMMA_BF16(ap.v, bvf[j].v, acc[j]);
    wave_lds_fence();   // before next tile rewrites Ssl/Psl
  };

  BF16x16 ka0[2], ka1[2], kb0[2], kb1[2];
  loadK(ka0, ka1, 0);
  for (int kt = 0; kt < T; kt += 64) {
    step(kt,      ka0, ka1, kb0, kb1, true);          // loads kt+32
    step(kt + 32, kb0, kb1, ka0, ka1, kt + 64 < T);   // loads kt+64
  }

  // ---- finalize: divide by row sum, store ctx bf16 (B,T,H*P) ----
  if (lane < 16) Csl[lane] = (l > 0.0f) ? (1.0f / l) : 0.0f;
  wave_lds_fence();
  float rin[8];
#pragma unroll
  for (int r = 0; r < 8; ++r) rin[r] = Csl[r + 8 * hl];
#pragma unroll
  for (int j = 0; j < 4; ++j)
#pragma unroll
    for (int r = 0; r < 8; ++r) {
      const int row = qt * 16 + r + 8 * hl;
      ctx[((size_t)b * T + row) * (H * P) + h * 64 + j * 16 + n] =
          f2bf(acc[j][r] * rin[r]);
    }
}

// ---------------------------------------------------------------------------
// Output projection: out(BT,D) = ctx(BT,HP) * Wo(HP,D) + bo. Double-buffered.
// ---------------------------------------------------------------------------
__global__ __launch_bounds__(256) void outproj_kernel(
    const __bf16* __restrict__ ctx,  // (B*T, HP) bf16
    const __bf16* __restrict__ WoT,  // (D, HP) bf16 pre-transposed
    const float* __restrict__ bo,    // (D)
    float* __restrict__ out) {       // (B*T, D) fp32
  const int D = 1024, HP = 1024;
  const int lane = threadIdx.x & 31;
  const int wid  = blockIdx.x * (blockDim.x >> 5) + (threadIdx.x >> 5);
  const int dTiles = D / 64;
  const int rowTile = wid / dTiles;
  const int dt = wid % dTiles;

  const int m  = lane & 15;
  const int hl = lane >> 4;
  const int n  = lane & 15;

  const __bf16* arow = ctx + (size_t)(rowTile * 16 + m) * HP;
  const __bf16* wn   = WoT + (size_t)(dt * 64 + n) * HP;

  auto loadA = [&](BF16x16& a, int kk) {
    a.h[0] = *(const v8bf*)(arow + kk + hl * 8);
    a.h[1] = *(const v8bf*)(arow + kk + 16 + hl * 8);
  };
  auto loadB = [&](BF16x16* dst, int kk) {
#pragma unroll
    for (int j = 0; j < 4; ++j) {
      const __bf16* wp = wn + (size_t)j * 16 * HP + kk + hl * 8;
      dst[j].h[0] = *(const v8bf*)(wp);
      dst[j].h[1] = *(const v8bf*)(wp + 16);
    }
  };

  v8f acc[4] = {};
  BF16x16 a0, a1, b0[4], b1[4];
  loadA(a0, 0);
  loadB(b0, 0);
  for (int kk = 0; kk < HP; kk += 64) {
    loadA(a1, kk + 32);
    loadB(b1, kk + 32);
#pragma unroll
    for (int j = 0; j < 4; ++j) acc[j] = WMMA_BF16(a0.v, b0[j].v, acc[j]);
    if (kk + 64 < HP) {
      loadA(a0, kk + 64);
      loadB(b0, kk + 64);
    }
#pragma unroll
    for (int j = 0; j < 4; ++j) acc[j] = WMMA_BF16(a1.v, b1[j].v, acc[j]);
  }

#pragma unroll
  for (int j = 0; j < 4; ++j) {
    const int dcol = dt * 64 + j * 16 + n;
    const float bv = bo[dcol];
#pragma unroll
    for (int r = 0; r < 8; ++r) {
      const int row = rowTile * 16 + r + 8 * hl;
      out[(size_t)row * D + dcol] = acc[j][r] + bv;
    }
  }
}

// ---------------------------------------------------------------------------
extern "C" void kernel_launch(void* const* d_in, const int* in_sizes, int n_in,
                              void* d_out, int out_size, void* d_ws, size_t ws_size,
                              hipStream_t stream) {
  const int B = 2, T = 2048, D = 1024, H = 16, P = 64, HP = H * P;

  const float* Q    = (const float*)d_in[0];
  const float* K    = (const float*)d_in[1];
  const float* V    = (const float*)d_in[2];
  const float* Wq   = (const float*)d_in[3];
  const float* bq   = (const float*)d_in[4];
  const float* Wk   = (const float*)d_in[5];
  const float* bk   = (const float*)d_in[6];
  const float* Wv   = (const float*)d_in[7];
  const float* bv   = (const float*)d_in[8];
  const float* Wo   = (const float*)d_in[9];
  const float* bo   = (const float*)d_in[10];
  const float* mask = (const float*)d_in[11];

  __bf16* ws = (__bf16*)d_ws;
  size_t off = 0;
  __bf16* WqT = ws + off; off += (size_t)H * P * D;      // (H,P,D)
  __bf16* WkT = ws + off; off += (size_t)H * P * D;
  __bf16* WvT = ws + off; off += (size_t)H * P * D;
  __bf16* WoT = ws + off; off += (size_t)D * HP;         // (D,HP)
  __bf16* qh  = ws + off; off += (size_t)B * H * T * P;  // (B,H,T,P)
  __bf16* kh  = ws + off; off += (size_t)B * H * T * P;
  __bf16* vhT = ws + off; off += (size_t)B * H * P * T;  // (B,H,P,T)
  __bf16* ctx = ws + off; off += (size_t)B * T * HP;     // (B,T,HP)

  // 1) transpose + bf16-convert weights
  {
    const int tw = H * D * P;
    transpose_w_kernel<<<(tw + 255) / 256, 256, 0, stream>>>(Wq, WqT, H, D, P);
    transpose_w_kernel<<<(tw + 255) / 256, 256, 0, stream>>>(Wk, WkT, H, D, P);
    transpose_w_kernel<<<(tw + 255) / 256, 256, 0, stream>>>(Wv, WvT, H, D, P);
    transpose_w_kernel<<<(HP * D + 255) / 256, 256, 0, stream>>>(Wo, WoT, 1, HP, D);
  }

  // 2) per-head projections (4096 waves -> 512 blocks x 8 waves)
  proj_kernel<0><<<512, 256, 0, stream>>>(Q, WqT, bq, qh);
  proj_kernel<0><<<512, 256, 0, stream>>>(K, WkT, bk, kh);
  proj_kernel<1><<<512, 256, 0, stream>>>(V, WvT, bv, vhT);

  // 3) flash attention: 4096 single-wave workgroups
  attn_kernel<<<4096, 32, 0, stream>>>(qh, kh, vhT, mask, ctx);

  // 4) output projection
  outproj_kernel<<<512, 256, 0, stream>>>(ctx, WoT, bo, (float*)d_out);
}
